// MyRNN_83253646065788
// MI455X (gfx1250) — matrix operand order repaired and checked by reading, not compiled, not run on hardware
//
#include <hip/hip_runtime.h>

// ---------------------------------------------------------------------------
// 2-layer tanh RNN (B=4096, T=512, H=32, input dim 1) + Linear(32,1) head.
// Strategy for MI455X (gfx1250, wave32, WMMA):
//   * one wave owns 16 batch rows; recurrence over T runs entirely on-chip
//   * per step: 6x v_wmma_f32_16x16x32_f16 (K=32 == H exactly), f32 accum
//   * weight B-fragments built once, held in VGPRs for all 512 steps
//   * D-layout -> A-layout conversion via padded LDS staging:
//       - K-permuted layout (slot s holds logical k=(s>>1)+16*(s&1)) so each
//         lane spills tile0/tile1 values as ONE packed ds_store_b32
//       - pack via single v_cvt_pk_rtz_f16_f32 per dword
//       - row stride 40 halves so half-wave groups hit disjoint bank groups
//   * layer-0 input term (rank-1, input dim 1) folded into C-init via bpermute
//   * tanh via native gfx1250 V_TANH_F32 (single trans op)
// ---------------------------------------------------------------------------

typedef __attribute__((ext_vector_type(16))) _Float16 v16h;
typedef __attribute__((ext_vector_type(8)))  _Float16 v8h;
typedef __attribute__((ext_vector_type(8)))  float    v8f;

#define B_TOT 4096
#define T_LEN 512
#define H_DIM 32
#define WPB   4            // waves per block
#define RPW   16           // batch rows per wave (WMMA M)
#define ROWP  40           // padded LDS row stride in halves (20 dwords)

union AFrag { v16h v; v8h h[2]; };

// K permutation: physical slot s (0..31) holds logical k.
__device__ __forceinline__ int kperm(int s) { return (s >> 1) + ((s & 1) << 4); }

// Native CDNA5 tanh (single VOP1 trans op) instead of ocml's branchy expansion.
__device__ __forceinline__ float fast_tanh(float x) {
#if __has_builtin(__builtin_amdgcn_tanhf)
    return __builtin_amdgcn_tanhf(x);
#elif __has_builtin(__builtin_amdgcn_tanh_f32)
    return __builtin_amdgcn_tanh_f32(x);
#else
    float r;
    asm("v_tanh_f32 %0, %1\n\tv_nop" : "=v"(r) : "v"(x));
    return r;
#endif
}

__device__ __forceinline__ v8f wmma_f16(v16h a, v16h b, v8f c) {
    // D = A(16x32 f16) * B(32x16 f16) + C(16x16 f32)
    return __builtin_amdgcn_wmma_f32_16x16x32_f16(
        /*neg_a=*/false, a, /*neg_b=*/false, b,
        /*c_mod=*/(short)0, c, /*reuse_a=*/false, /*reuse_b=*/false);
}

// One-instruction pack: v_cvt_pk_rtz_f16_f32 (two f32 -> packed 2xf16 dword).
__device__ __forceinline__ unsigned pack_f16x2(float lo, float hi) {
#if __has_builtin(__builtin_amdgcn_cvt_pkrtz)
    return __builtin_bit_cast(unsigned, __builtin_amdgcn_cvt_pkrtz(lo, hi));
#else
    unsigned r;
    asm("v_cvt_pk_rtz_f16_f32 %0, %1, %2" : "=v"(r) : "v"(lo), "v"(hi));
    return r;
#endif
}

// B fragment for h @ W^T with K-permutation: element e is physical slot
// s = 8*(lane>=16) + e + 8*(e>=8), logical k = kperm(s);  B[k][n] = W[(tb+n)*32 + k].
__device__ __forceinline__ v16h make_bfrag(const float* __restrict__ W,
                                           int tb, int lane) {
    const int n    = lane & 15;
    const int soff = (lane >> 4) << 3;            // 0 or 8
    const float* row = W + (tb + n) * H_DIM;
    v16h f;
#pragma unroll
    for (int e = 0; e < 16; ++e) f[e] = (_Float16)row[kperm(soff + e + ((e >> 3) << 3))];
    return f;
}

// A fragment from the padded K-permuted LDS tile: lanes 0-15 hold physical
// slots 0..7 & 16..23, lanes 16-31 hold 8..15 & 24..31 -> two b128 loads.
__device__ __forceinline__ v16h load_afrag(const _Float16* __restrict__ hb,
                                           int lane) {
    const int m  = lane & 15;
    const int s0 = (lane >> 4) << 3;              // 0 or 8
    AFrag a;
    a.h[0] = *(const v8h*)(hb + m * ROWP + s0);
    a.h[1] = *(const v8h*)(hb + m * ROWP + 16 + s0);
    return a.v;
}

// Initial A fragment straight from global f32 h_state[layer][b][32],
// applying the same K-permutation.
__device__ __forceinline__ v16h load_afrag_global(const float* __restrict__ hs,
                                                  int bbase, int lane) {
    const int m  = lane & 15;
    const int hi = lane >> 4;
    const float* p = hs + (size_t)(bbase + m) * H_DIM;
    v16h f;
#pragma unroll
    for (int e = 0; e < 16; ++e) {
        const int s = (e < 8) ? (8 * hi + e) : (16 + 8 * hi + (e - 8));
        f[e] = (_Float16)p[kperm(s)];
    }
    return f;
}

// Spill one 16x32 h tile (two 16x16 f32 D tiles) to the K-permuted LDS tile.
// Logical j=n -> slot 2n (dword n, lo half); j=16+n -> slot 2n+1 (hi half):
// one v_cvt_pk_rtz + one ds_store_b32 per (lane,r).
__device__ __forceinline__ void store_h(_Float16* __restrict__ hb,
                                        const v8f& ca, const v8f& cb, int lane) {
    const int n  = lane & 15;
    const int mb = (lane >> 4) << 3;
    unsigned* dw = (unsigned*)hb;
#pragma unroll
    for (int r = 0; r < 8; ++r)
        dw[(mb + r) * (ROWP / 2) + n] = pack_f16x2(ca[r], cb[r]);
}

__global__ __launch_bounds__(WPB * 32)
void rnn2_wmma_kernel(const float* __restrict__ x,        // [B,T,1]
                      const float* __restrict__ h_state,  // [2,B,32]
                      const float* __restrict__ Wih0,     // [32,1]
                      const float* __restrict__ Whh0,     // [32,32]
                      const float* __restrict__ bih0,
                      const float* __restrict__ bhh0,
                      const float* __restrict__ Wih1,     // [32,32]
                      const float* __restrict__ Whh1,     // [32,32]
                      const float* __restrict__ bih1,
                      const float* __restrict__ bhh1,
                      const float* __restrict__ Wfc,      // [1,32]
                      const float* __restrict__ bfc,      // [1]
                      float* __restrict__ out)            // [B pred][2*B*32 h]
{
    __shared__ __align__(128) _Float16 hbuf[WPB][2][RPW][ROWP]; // 10 KB
    __shared__ __align__(128) float    fbuf[WPB][RPW][H_DIM];   // 8 KB

    const int lane  = threadIdx.x & 31;
    const int w     = threadIdx.x >> 5;
    const int wave  = blockIdx.x * WPB + w;
    const int bbase = wave * RPW;

    const int n  = lane & 15;        // WMMA N / column for C,D and B frags
    const int mb = (lane >> 4) << 3; // C-row base (0 or 8)

    // per-lane constants (depend only on n): biases and rank-1 input weights
    const float wv0 = Wih0[n];
    const float wv1 = Wih0[16 + n];
    const float b0a = bih0[n]      + bhh0[n];
    const float b0b = bih0[16 + n] + bhh0[16 + n];
    const float b1a = bih1[n]      + bhh1[n];
    const float b1b = bih1[16 + n] + bhh1[16 + n];

    // time-invariant weight B fragments, resident in VGPRs for all 512 steps
    const v16h bWhh0_0 = make_bfrag(Whh0, 0, lane);
    const v16h bWhh0_1 = make_bfrag(Whh0, 16, lane);
    const v16h bWih1_0 = make_bfrag(Wih1, 0, lane);
    const v16h bWih1_1 = make_bfrag(Wih1, 16, lane);
    const v16h bWhh1_0 = make_bfrag(Whh1, 0, lane);
    const v16h bWhh1_1 = make_bfrag(Whh1, 16, lane);

    // initial hidden states
    v16h a0 = load_afrag_global(h_state, bbase, lane);
    v16h a1 = load_afrag_global(h_state + (size_t)B_TOT * H_DIM, bbase, lane);

    // lane l sources x for batch row (bbase + l&15); bpermute broadcasts it
    const float* xrow = x + (size_t)(bbase + n) * T_LEN;

    v8f c0a, c0b, c1a, c1b;

    for (int t = 0; t < T_LEN; ++t) {
        const float xv = xrow[t];

        // ---- layer 0: C init = bias + x * wih0 (rank-1 input projection)
#pragma unroll
        for (int r = 0; r < 8; ++r) {
            const float xm = __shfl(xv, mb + r, 32);   // row M = mb+r
            c0a[r] = fmaf(xm, wv0, b0a);
            c0b[r] = fmaf(xm, wv1, b0b);
        }
        c0a = wmma_f16(a0, bWhh0_0, c0a);
        c0b = wmma_f16(a0, bWhh0_1, c0b);
#pragma unroll
        for (int r = 0; r < 8; ++r) {
            c0a[r] = fast_tanh(c0a[r]);
            c0b[r] = fast_tanh(c0b[r]);
        }
        store_h(&hbuf[w][0][0][0], c0a, c0b, lane);
        asm volatile("s_wait_dscnt 0" ::: "memory");
        a0 = load_afrag(&hbuf[w][0][0][0], lane);      // = ys0_t, A layout

        // ---- layer 1
#pragma unroll
        for (int r = 0; r < 8; ++r) { c1a[r] = b1a; c1b[r] = b1b; }
        c1a = wmma_f16(a0, bWih1_0, c1a);
        c1b = wmma_f16(a0, bWih1_1, c1b);
        c1a = wmma_f16(a1, bWhh1_0, c1a);
        c1b = wmma_f16(a1, bWhh1_1, c1b);
#pragma unroll
        for (int r = 0; r < 8; ++r) {
            c1a[r] = fast_tanh(c1a[r]);
            c1b[r] = fast_tanh(c1b[r]);
        }
        store_h(&hbuf[w][1][0][0], c1a, c1b, lane);
        asm volatile("s_wait_dscnt 0" ::: "memory");
        a1 = load_afrag(&hbuf[w][1][0][0], lane);
    }

    // ---- h_new outputs (f32, from final tanh'd accumulators)
    float* hout0 = out + B_TOT;
    float* hout1 = out + B_TOT + (size_t)B_TOT * H_DIM;
#pragma unroll
    for (int r = 0; r < 8; ++r) {
        const int m = mb + r;
        hout0[(size_t)(bbase + m) * H_DIM + n]      = c0a[r];
        hout0[(size_t)(bbase + m) * H_DIM + 16 + n] = c0b[r];
        hout1[(size_t)(bbase + m) * H_DIM + n]      = c1a[r];
        hout1[(size_t)(bbase + m) * H_DIM + 16 + n] = c1b[r];
    }

    // ---- pred = last_h1 @ Wfc^T + bfc  (one-time 32-wide dot per row)
#pragma unroll
    for (int r = 0; r < 8; ++r) {
        fbuf[w][mb + r][n]      = c1a[r];
        fbuf[w][mb + r][16 + n] = c1b[r];
    }
    asm volatile("s_wait_dscnt 0" ::: "memory");
    if (lane < 16) {
        float s = bfc[0];
#pragma unroll
        for (int j = 0; j < H_DIM; ++j)
            s = fmaf(fbuf[w][lane][j], Wfc[j], s);
        out[bbase + lane] = s;
    }
}

extern "C" void kernel_launch(void* const* d_in, const int* in_sizes, int n_in,
                              void* d_out, int out_size, void* d_ws, size_t ws_size,
                              hipStream_t stream) {
    (void)in_sizes; (void)n_in; (void)out_size; (void)d_ws; (void)ws_size;
    const float* x       = (const float*)d_in[0];
    const float* h_state = (const float*)d_in[1];
    const float* Wih0    = (const float*)d_in[2];
    const float* Whh0    = (const float*)d_in[3];
    const float* bih0    = (const float*)d_in[4];
    const float* bhh0    = (const float*)d_in[5];
    const float* Wih1    = (const float*)d_in[6];
    const float* Whh1    = (const float*)d_in[7];
    const float* bih1    = (const float*)d_in[8];
    const float* bhh1    = (const float*)d_in[9];
    const float* Wfc     = (const float*)d_in[10];
    const float* bfc     = (const float*)d_in[11];
    float* out = (float*)d_out;

    dim3 grid(B_TOT / (WPB * RPW));   // 4096 / 64 = 64 blocks
    dim3 block(WPB * 32);             // 4 waves of 32
    rnn2_wmma_kernel<<<grid, block, 0, stream>>>(
        x, h_state, Wih0, Whh0, bih0, bhh0,
        Wih1, Whh1, bih1, bhh1, Wfc, bfc, out);
}
